// MultiHeadAttention_26542897889650
// MI455X (gfx1250) — matrix operand order repaired and checked by reading, not compiled, not run on hardware
//
#include <hip/hip_runtime.h>
#include <hip/hip_bf16.h>

#define B_   4
#define S_   2048
#define D_   1024
#define H_   16
#define DH_  64
#define HID_ 64
#define NEGBIG 1e18f

typedef __bf16 v16bf __attribute__((ext_vector_type(16)));
typedef __bf16 v2bf  __attribute__((ext_vector_type(2)));
typedef float  v8f   __attribute__((ext_vector_type(8)));
typedef int    v4ig  __attribute__((vector_size(16)));   // gcc-style int4, matches builtin proto

union Frag16 { v16bf v; unsigned u[8]; };

#if __has_builtin(__builtin_amdgcn_global_load_async_to_lds_b128)
#define HAVE_ASYNC_LDS 1
#else
#define HAVE_ASYNC_LDS 0
#endif

#if HAVE_ASYNC_LDS
__device__ __forceinline__ void async_copy16(void* lds_dst, const void* gsrc) {
  __builtin_amdgcn_global_load_async_to_lds_b128(
      (__attribute__((address_space(1))) v4ig*)gsrc,
      (__attribute__((address_space(3))) v4ig*)lds_dst, 0, 0);
}
__device__ __forceinline__ void wait_async0() {
#if __has_builtin(__builtin_amdgcn_s_wait_asynccnt)
  __builtin_amdgcn_s_wait_asynccnt(0);
#else
  asm volatile("s_wait_asynccnt 0" ::: "memory");
#endif
}
#endif

// Native f32 -> bf16 (compiler picks hardware cvt; RNE per ISA)
__device__ __forceinline__ unsigned short f2bf(float f) {
  __bf16 h = (__bf16)f;
  return __builtin_bit_cast(unsigned short, h);
}
__device__ __forceinline__ unsigned pack2(float a, float b) {
  v2bf t;
  t[0] = (__bf16)a;
  t[1] = (__bf16)b;
  return __builtin_bit_cast(unsigned, t);
}
// 16-bit A-matrix (16x32) per-lane K pair base: VGPR v, lane half
__device__ __forceinline__ int kbA(int v, int half) {
  return (v < 4) ? (2 * v + 8 * half) : (16 + 2 * (v - 4) + 8 * half);
}
// 16-bit B-matrix (32x16) per-lane K pair base
__device__ __forceinline__ int kbB(int v, int half) {
  return (v < 4) ? (2 * v + 8 * half) : (16 + 2 * (v - 4) + 8 * half);
}
__device__ __forceinline__ v8f zero8() {
  v8f z = {0.f, 0.f, 0.f, 0.f, 0.f, 0.f, 0.f, 0.f};
  return z;
}

// dst[b][c][r] = bf16(src[b][r][c])
__global__ void convert_T(const float* __restrict__ src, unsigned short* __restrict__ dst,
                          int batch, int R, int C) {
  int i = blockIdx.x * blockDim.x + threadIdx.x;
  int total = batch * R * C;
  if (i >= total) return;
  int c = i % C;
  int t = i / C;
  int r = t % R;
  int b = t / R;
  dst[(b * C + c) * R + r] = f2bf(src[i]);
}

// Per-head Dense projections: qh/kh [bh][s][e] bf16, v stored transposed vT[bh][e][s]
__global__ __launch_bounds__(32) void proj_kernel(
    const float* __restrict__ q, const float* __restrict__ k, const float* __restrict__ v,
    const unsigned short* __restrict__ WqT, const unsigned short* __restrict__ WkT,
    const unsigned short* __restrict__ WvT,
    const float* __restrict__ bq, const float* __restrict__ bk, const float* __restrict__ bv,
    unsigned short* __restrict__ qh, unsigned short* __restrict__ kh,
    unsigned short* __restrict__ vT) {
  const int lane = threadIdx.x & 31;
  const int half = lane >> 4;
  const int l16  = lane & 15;
  const int mbase = blockIdx.x * 16;
  const int bh = blockIdx.y;
  const int b  = bh / H_;
  const int h  = bh % H_;
  const int which = blockIdx.z;

  const float* src = (which == 0) ? q : (which == 1) ? k : v;
  const unsigned short* WT = (which == 0) ? WqT : (which == 1) ? WkT : WvT;
  const float* bias = (which == 0) ? bq : (which == 1) ? bk : bv;

  // A fragments: rows = 16 sequence positions, K = d (64 -> two chunks of 32)
  Frag16 A[2];
  const float* srow = src + (b * S_ + mbase + l16) * D_ + h * DH_;
#pragma unroll
  for (int kk = 0; kk < 2; ++kk) {
#pragma unroll
    for (int vv = 0; vv < 8; ++vv) {
      int d = kk * 32 + kbA(vv, half);
      A[kk].u[vv] = pack2(srow[d], srow[d + 1]);
    }
  }

  const unsigned short* WTh = WT + h * HID_ * DH_;  // [j][d] bf16
#pragma unroll
  for (int nt = 0; nt < 4; ++nt) {
    v8f c = zero8();
#pragma unroll
    for (int kk = 0; kk < 2; ++kk) {
      Frag16 Bf;
#pragma unroll
      for (int vv = 0; vv < 8; ++vv) {
        int d = kk * 32 + kbB(vv, half);
        Bf.u[vv] = *(const unsigned*)&WTh[(nt * 16 + l16) * DH_ + d];
      }
      c = __builtin_amdgcn_wmma_f32_16x16x32_bf16(false, A[kk].v, false, Bf.v,
                                                  (short)0, c, false, false);
    }
    float bj = bias[h * HID_ + nt * 16 + l16];
    if (which == 2) {
      unsigned short* dst = vT + bh * HID_ * S_;
#pragma unroll
      for (int r = 0; r < 8; ++r) {
        int s_i = mbase + r + 8 * half;
        dst[(nt * 16 + l16) * S_ + s_i] = f2bf(c[r] + bj);
      }
    } else {
      unsigned short* dst = ((which == 0) ? qh : kh) + bh * S_ * HID_;
#pragma unroll
      for (int r = 0; r < 8; ++r) {
        int s_i = mbase + r + 8 * half;
        dst[s_i * HID_ + nt * 16 + l16] = f2bf(c[r] + bj);
      }
    }
  }
}

// Flash attention: block = 4 waves x 16 query rows; key tiles of 32.
__global__ __launch_bounds__(128) void attn_kernel(
    const unsigned short* __restrict__ qh, const unsigned short* __restrict__ kh,
    const unsigned short* __restrict__ vT, const int* __restrict__ key_mask,
    unsigned short* __restrict__ concat) {
  __shared__ unsigned short lds_k[32 * 64];   // [t][e]
  __shared__ unsigned short lds_v[64 * 32];   // [e][t]
  __shared__ unsigned short lds_p[4 * 16 * 32];

  const int tid  = threadIdx.x;
  const int w    = tid >> 5;
  const int lane = tid & 31;
  const int half = lane >> 4;
  const int l16  = lane & 15;
  const int qb0  = blockIdx.x * 64;
  const int bh   = blockIdx.y;
  const int b    = bh / H_;
  const int h    = bh % H_;
  const int qw   = qb0 + w * 16;

  const unsigned short* qhp = qh + bh * S_ * HID_;
  const unsigned short* khp = kh + bh * S_ * HID_;
  const unsigned short* vTp = vT + bh * HID_ * S_;
  const int* kmp = key_mask + b * S_;

  // Q A-fragments (e = 0..31, 32..63)
  Frag16 Aq[2];
#pragma unroll
  for (int kk = 0; kk < 2; ++kk)
#pragma unroll
    for (int vv = 0; vv < 8; ++vv)
      Aq[kk].u[vv] = *(const unsigned*)&qhp[(qw + l16) * HID_ + kk * 32 + kbA(vv, half)];

  v8f o[4];
#pragma unroll
  for (int f = 0; f < 4; ++f) o[f] = zero8();
  float mrow[8], lrow[8];
#pragma unroll
  for (int r = 0; r < 8; ++r) { mrow[r] = -3.0e38f; lrow[r] = 0.f; }

  const int kend = qb0 + 64;  // causal upper bound for this block
  for (int kt = 0; kt < kend; kt += 32) {
    // stage K tile (32 rows x 64 e, contiguous) and V tile (64 e x 32 t)
    {
      const uint4* gk = (const uint4*)(khp + kt * HID_);
      uint4* sk = (uint4*)lds_k;
      int e = tid >> 1, hf = tid & 1;
      const uint4* gv = (const uint4*)(vTp + e * S_ + kt + hf * 16);
      uint4* sv = (uint4*)(lds_v + e * 32 + hf * 16);
#if HAVE_ASYNC_LDS
      async_copy16(sk + tid,       gk + tid);
      async_copy16(sk + tid + 128, gk + tid + 128);
      async_copy16(sv,             gv);
      async_copy16(sv + 1,         gv + 1);
#else
      sk[tid]       = gk[tid];
      sk[tid + 128] = gk[tid + 128];
      sv[0] = gv[0];
      sv[1] = gv[1];
#endif
      if (kt + 32 < kend) __builtin_prefetch(khp + (kt + 32) * HID_ + tid * 16, 0, 3);
    }
#if HAVE_ASYNC_LDS
    wait_async0();
#endif
    __syncthreads();

    if (kt < qw + 16) {  // wave-uniform causal skip
      float sc[2][8];
#pragma unroll
      for (int tt = 0; tt < 2; ++tt) {
        v8f c = zero8();
#pragma unroll
        for (int kk = 0; kk < 2; ++kk) {
          Frag16 Bk;
#pragma unroll
          for (int vv = 0; vv < 8; ++vv)
            Bk.u[vv] = *(const unsigned*)&lds_k[(tt * 16 + l16) * 64 + kk * 32 + kbB(vv, half)];
          c = __builtin_amdgcn_wmma_f32_16x16x32_bf16(false, Aq[kk].v, false, Bk.v,
                                                      (short)0, c, false, false);
        }
        int tg = kt + tt * 16 + l16;
        float pen = (kmp[tg] == 0) ? NEGBIG : 0.0f;
#pragma unroll
        for (int r = 0; r < 8; ++r) {
          int row = qw + r + 8 * half;
          sc[tt][r] = c[r] - pen - ((tg > row) ? NEGBIG : 0.f);
        }
      }
      // online softmax (row reductions across 16 lanes of each half)
#pragma unroll
      for (int r = 0; r < 8; ++r) {
        float x = fmaxf(sc[0][r], sc[1][r]);
#pragma unroll
        for (int d = 1; d < 16; d <<= 1) x = fmaxf(x, __shfl_xor(x, d, 16));
        float mi = fmaxf(mrow[r], x);
        float scale = __expf(mrow[r] - mi);
        float p0 = __expf(sc[0][r] - mi);
        float p1 = __expf(sc[1][r] - mi);
        float sum = p0 + p1;
#pragma unroll
        for (int d = 1; d < 16; d <<= 1) sum += __shfl_xor(sum, d, 16);
        lrow[r] = lrow[r] * scale + sum;
        mrow[r] = mi;
#pragma unroll
        for (int f = 0; f < 4; ++f) o[f][r] = o[f][r] * scale;
        // P tile to LDS (C-layout -> row-major bf16)
        lds_p[w * 512 + (r + 8 * half) * 32 + l16]      = f2bf(p0);
        lds_p[w * 512 + (r + 8 * half) * 32 + 16 + l16] = f2bf(p1);
      }
      asm volatile("" ::: "memory");  // keep P stores before A-fragment reload
      Frag16 Pa;
#pragma unroll
      for (int vv = 0; vv < 8; ++vv)
        Pa.u[vv] = *(const unsigned*)&lds_p[w * 512 + l16 * 32 + kbA(vv, half)];
#pragma unroll
      for (int f = 0; f < 4; ++f) {
        Frag16 Bv;
#pragma unroll
        for (int vv = 0; vv < 8; ++vv)
          Bv.u[vv] = *(const unsigned*)&lds_v[(f * 16 + l16) * 32 + kbB(vv, half)];
        o[f] = __builtin_amdgcn_wmma_f32_16x16x32_bf16(false, Pa.v, false, Bv.v,
                                                       (short)0, o[f], false, false);
      }
    }
    __syncthreads();
  }

  unsigned short* cp = concat + b * S_ * (H_ * HID_);
#pragma unroll
  for (int r = 0; r < 8; ++r) {
    float inv = 1.0f / lrow[r];
    int row = qw + r + 8 * half;
#pragma unroll
    for (int f = 0; f < 4; ++f)
      cp[row * (H_ * HID_) + h * HID_ + f * 16 + l16] = f2bf(o[f][r] * inv);
  }
}

// out[M=8192][1024] = concat_bf16 @ Wo + bo
__global__ __launch_bounds__(128) void outproj_kernel(
    const unsigned short* __restrict__ concat, const unsigned short* __restrict__ WoT,
    const float* __restrict__ bo, float* __restrict__ out) {
  const int tid = threadIdx.x, w = tid >> 5, lane = tid & 31;
  const int half = lane >> 4, l16 = lane & 15;
  const int mb = blockIdx.x * 64 + w * 16;
  const int nb = blockIdx.y * 64;
  v8f c[4];
#pragma unroll
  for (int f = 0; f < 4; ++f) c[f] = zero8();
#pragma unroll 2
  for (int k0 = 0; k0 < D_; k0 += 32) {
    Frag16 Af;
#pragma unroll
    for (int vv = 0; vv < 8; ++vv)
      Af.u[vv] = *(const unsigned*)&concat[(mb + l16) * D_ + k0 + kbA(vv, half)];
#pragma unroll
    for (int f = 0; f < 4; ++f) {
      Frag16 Bf;
#pragma unroll
      for (int vv = 0; vv < 8; ++vv)
        Bf.u[vv] = *(const unsigned*)&WoT[(nb + f * 16 + l16) * D_ + k0 + kbB(vv, half)];
      c[f] = __builtin_amdgcn_wmma_f32_16x16x32_bf16(false, Af.v, false, Bf.v,
                                                     (short)0, c[f], false, false);
    }
  }
#pragma unroll
  for (int f = 0; f < 4; ++f) {
    float bj = bo[nb + f * 16 + l16];
#pragma unroll
    for (int r = 0; r < 8; ++r)
      out[(mb + r + 8 * half) * D_ + nb + f * 16 + l16] = c[f][r] + bj;
  }
}

extern "C" void kernel_launch(void* const* d_in, const int* in_sizes, int n_in,
                              void* d_out, int out_size, void* d_ws, size_t ws_size,
                              hipStream_t stream) {
  (void)in_sizes; (void)n_in; (void)out_size; (void)ws_size;
  const float* q  = (const float*)d_in[0];
  const float* k  = (const float*)d_in[1];
  const float* v  = (const float*)d_in[2];
  const int*   km = (const int*)d_in[3];
  const float* Wq = (const float*)d_in[4];
  const float* bq = (const float*)d_in[5];
  const float* Wk = (const float*)d_in[6];
  const float* bk = (const float*)d_in[7];
  const float* Wv = (const float*)d_in[8];
  const float* bv = (const float*)d_in[9];
  const float* Wo = (const float*)d_in[10];
  const float* bo = (const float*)d_in[11];

  char* ws = (char*)d_ws;
  unsigned short* WqT = (unsigned short*)(ws + 0);
  unsigned short* WkT = (unsigned short*)(ws + 131072);
  unsigned short* WvT = (unsigned short*)(ws + 262144);
  unsigned short* WoT = (unsigned short*)(ws + 393216);
  unsigned short* qh  = (unsigned short*)(ws + 2490368);
  unsigned short* kh  = (unsigned short*)(ws + 19267584);
  unsigned short* vT  = (unsigned short*)(ws + 36044800);
  unsigned short* cc  = (unsigned short*)(ws + 52822016);
  float* out = (float*)d_out;

  int nW = H_ * DH_ * HID_;
  convert_T<<<(nW + 255) / 256, 256, 0, stream>>>(Wq, WqT, H_, DH_, HID_);
  convert_T<<<(nW + 255) / 256, 256, 0, stream>>>(Wk, WkT, H_, DH_, HID_);
  convert_T<<<(nW + 255) / 256, 256, 0, stream>>>(Wv, WvT, H_, DH_, HID_);
  convert_T<<<(D_ * D_ + 255) / 256, 256, 0, stream>>>(Wo, WoT, 1, D_, D_);

  dim3 pg(S_ / 16, B_ * H_, 3);
  proj_kernel<<<pg, 32, 0, stream>>>(q, k, v, WqT, WkT, WvT, bq, bk, bv, qh, kh, vT);

  dim3 ag(S_ / 64, B_ * H_);
  attn_kernel<<<ag, 128, 0, stream>>>(qh, kh, vT, km, cc);

  dim3 og((B_ * S_) / 64, D_ / 64);
  outproj_kernel<<<og, 128, 0, stream>>>(cc, WoT, bo, out);
}
